// ConfounderStackLayers_44925357916624
// MI455X (gfx1250) — matrix operand order repaired and checked by reading, not compiled, not run on hardware
//
#include <hip/hip_runtime.h>

// ---------------------------------------------------------------------------
// Grouped 3-layer MLP (C=8 confounders), fully fused per 128-row batch tile.
// Round 2: one-shot f32->bf16 pre-pass into d_ws (fallback: in-kernel cvt),
// layer-0 A fragments hoisted into registers, bf16 WMMA w/ f32 accumulation.
// ---------------------------------------------------------------------------

typedef __attribute__((ext_vector_type(16))) __bf16 bf16x16;
typedef __attribute__((ext_vector_type(4)))  __bf16 bf16x4;
typedef __attribute__((ext_vector_type(8)))  float  f32x8;

struct U256 { uint4 lo; uint4 hi; };   // 32 bytes == one bf16x16 fragment

constexpr int kBatch   = 16384;
constexpr int kC       = 8;
constexpr int kD       = 256;
constexpr int kH0      = 512;
constexpr int kH1      = 256;
constexpr int kMT      = 128;     // batch rows per workgroup
constexpr int kThreads = 256;     // 8 wave32 waves

// LDS partition (bf16 elements). Total = 147456 elems = 288 KB.
constexpr int XS_OFF = 0;                   // x tile   [kMT][kD]
constexpr int HS_OFF = XS_OFF + kMT * kD;   // relu(h0) [kMT][kH0]
constexpr int H1_OFF = HS_OFF + kMT * kH0;  // relu(h1) [kMT][kH1]
constexpr int WS_OFF = H1_OFF + kMT * kH1;  // W stage  [64][256]
constexpr int SMEM_ELEMS = WS_OFF + 64 * 256;

// Workspace layout for the bf16 pre-pass (element counts).
constexpr size_t XB_ELEMS  = (size_t)kBatch * kD;     // 4 Mi
constexpr size_t W0B_ELEMS = (size_t)kC * kH0 * kD;   // 1 Mi
constexpr size_t W1B_ELEMS = (size_t)kC * kH1 * kH0;  // 1 Mi
constexpr size_t WS_NEEDED = (XB_ELEMS + W0B_ELEMS + W1B_ELEMS) * 2;  // 12.6 MB

__device__ __forceinline__ __bf16 f2bf(float f) {
  // round-to-nearest-even f32 -> bf16
  unsigned u = __builtin_bit_cast(unsigned, f);
  unsigned r = (u + 0x7FFFu + ((u >> 16) & 1u)) >> 16;
  return __builtin_bit_cast(__bf16, (unsigned short)r);
}

// A fragment (16x32 bf16, M x K) from row-major [M][>=K] storage.
// Lane (m = lane&15, g = lane>>4) holds k in {g*8..g*8+7} and {16+g*8..+7}.
__device__ __forceinline__ bf16x16 load_fragA(const __bf16* base, int stride, int lane) {
  const __bf16* p = base + (lane & 15) * stride + ((lane >> 4) << 3);
  U256 t;
  t.lo = *reinterpret_cast<const uint4*>(p);
  t.hi = *reinterpret_cast<const uint4*>(p + 16);
  return __builtin_bit_cast(bf16x16, t);
}

// B fragment (32x16 bf16, K x N) from row-major [N][>=K] storage (== B^T, the
// natural [out][in] weight layout). Lane (n = lane&15, g = lane>>4) holds
// k in {g*8..g*8+7} and {16+g*8..+7}.
__device__ __forceinline__ bf16x16 load_fragB(const __bf16* base, int stride, int lane) {
  const __bf16* p = base + (lane & 15) * stride + ((lane >> 4) << 3);
  U256 t;
  t.lo = *reinterpret_cast<const uint4*>(p);
  t.hi = *reinterpret_cast<const uint4*>(p + 16);
  return __builtin_bit_cast(bf16x16, t);
}

// ---------------- pre-pass: bulk f32 -> bf16 conversion ----------------
__global__ __launch_bounds__(256)
void cvt_f32_to_bf16(const float* __restrict__ src, __bf16* __restrict__ dst, int n4) {
  int i = blockIdx.x * blockDim.x + threadIdx.x;
  if (i < n4) {
    float4 v = reinterpret_cast<const float4*>(src)[i];
    bf16x4 o = { f2bf(v.x), f2bf(v.y), f2bf(v.z), f2bf(v.w) };
    reinterpret_cast<bf16x4*>(dst)[i] = o;
  }
}

// ---------------- fused grouped-MLP kernel ----------------
template<bool BF16IN>
__global__ __launch_bounds__(kThreads)
void ConfounderStackLayers_kernel(const void* __restrict__ xin,
                                  const void* __restrict__ w0in, const float* __restrict__ b0,
                                  const void* __restrict__ w1in, const float* __restrict__ b1,
                                  const float* __restrict__ W2,  const float* __restrict__ b2,
                                  float* __restrict__ out)
{
  extern __shared__ __bf16 smem[];
  __bf16* xs  = smem + XS_OFF;
  __bf16* hs  = smem + HS_OFF;
  __bf16* h1s = smem + H1_OFF;
  __bf16* ws  = smem + WS_OFF;

  const int tid  = threadIdx.x;
  const int lane = tid & 31;
  const int r0   = (tid >> 5) * 16;        // this wave's 16-row strip
  const int c    = blockIdx.y;             // confounder index
  const int row0 = blockIdx.x * kMT;       // batch-tile origin

  // ---- Stage x tile into LDS (row-major [kMT][kD] bf16) ----
  if constexpr (BF16IN) {
    const uint4* xg = reinterpret_cast<const uint4*>(
        (const __bf16*)xin + (size_t)row0 * kD);
    #pragma unroll 4
    for (int i = tid; i < kMT * kD / 8; i += kThreads)
      *reinterpret_cast<uint4*>(xs + i * 8) = xg[i];
  } else {
    const float4* xg = reinterpret_cast<const float4*>(
        (const float*)xin + (size_t)row0 * kD);
    #pragma unroll 4
    for (int i = tid; i < kMT * kD / 4; i += kThreads) {
      float4 v = xg[i];
      bf16x4 o = { f2bf(v.x), f2bf(v.y), f2bf(v.z), f2bf(v.w) };
      *reinterpret_cast<bf16x4*>(xs + i * 4) = o;
    }
  }
  __syncthreads();

  // Hoist all layer-0 A fragments (16 rows x K=256) into registers: 64 VGPRs,
  // reused across all 8 N-chunks (was 128 redundant ds_load_b128 per wave).
  bf16x16 afr[8];
  #pragma unroll
  for (int ks = 0; ks < 8; ++ks)
    afr[ks] = load_fragA(xs + r0 * kD + ks * 32, kD, lane);

  // ================= Layer 0:  h0 = relu(x @ W0[c]^T + b0[c]) =================
  const float* b0g = b0 + c * kH0;
  for (int nc = 0; nc < kH0 / 64; ++nc) {          // 8 chunks of 64 output cols
    if constexpr (BF16IN) {
      const uint4* wg = reinterpret_cast<const uint4*>(
          (const __bf16*)w0in + ((size_t)c * kH0 + nc * 64) * kD);
      #pragma unroll 4
      for (int i = tid; i < 64 * kD / 8; i += kThreads)
        *reinterpret_cast<uint4*>(ws + i * 8) = wg[i];
    } else {
      const float4* wg = reinterpret_cast<const float4*>(
          (const float*)w0in + ((size_t)c * kH0 + nc * 64) * kD);
      #pragma unroll 4
      for (int i = tid; i < 64 * kD / 4; i += kThreads) {
        float4 v = wg[i];
        bf16x4 o = { f2bf(v.x), f2bf(v.y), f2bf(v.z), f2bf(v.w) };
        *reinterpret_cast<bf16x4*>(ws + i * 4) = o;
      }
    }
    __syncthreads();

    f32x8 acc[4] = {};
    #pragma unroll
    for (int ks = 0; ks < kD / 32; ++ks) {
      #pragma unroll
      for (int nt = 0; nt < 4; ++nt) {
        bf16x16 bm = load_fragB(ws + nt * 16 * kD + ks * 32, kD, lane);
        acc[nt] = __builtin_amdgcn_wmma_f32_16x16x32_bf16(
            false, afr[ks], false, bm, (short)0, acc[nt], false, false);
      }
    }
    // bias + relu + pack to bf16 LDS (C/D layout: M = v + 8*(lane>>4), N = lane&15)
    const int mrow = r0 + ((lane >> 4) << 3);
    #pragma unroll
    for (int nt = 0; nt < 4; ++nt) {
      const int col  = nc * 64 + nt * 16 + (lane & 15);
      const float bias = b0g[col];
      #pragma unroll
      for (int v = 0; v < 8; ++v) {
        float val = acc[nt][v] + bias;
        hs[(mrow + v) * kH0 + col] = f2bf(val > 0.f ? val : 0.f);
      }
    }
    __syncthreads();   // ws reuse + hs completion
  }

  // ================= Layer 1:  h1 = relu(h0 @ W1[c]^T + b1[c]) =================
  const float* b1g = b1 + c * kH1;
  for (int nc = 0; nc < kH1 / 64; ++nc) {          // 4 chunks of 64 output cols
    f32x8 acc[4] = {};
    for (int kt = 0; kt < kH0 / 256; ++kt) {       // 2 K-tiles of 256
      if constexpr (BF16IN) {
        const __bf16* wb = (const __bf16*)w1in +
            ((size_t)c * kH1 + nc * 64) * kH0 + kt * 256;
        #pragma unroll 4
        for (int i = tid; i < 64 * 32; i += kThreads) {   // 64 rows x 32 uint4
          const int n = i >> 5, d8 = i & 31;
          reinterpret_cast<uint4*>(ws + n * 256)[d8] =
              reinterpret_cast<const uint4*>(wb + (size_t)n * kH0)[d8];
        }
      } else {
        const float* wb = (const float*)w1in +
            ((size_t)c * kH1 + nc * 64) * kH0 + kt * 256;
        #pragma unroll 4
        for (int i = tid; i < 64 * 64; i += kThreads) {
          const int n = i >> 6, d4 = i & 63;
          float4 v = reinterpret_cast<const float4*>(wb + (size_t)n * kH0)[d4];
          bf16x4 o = { f2bf(v.x), f2bf(v.y), f2bf(v.z), f2bf(v.w) };
          *reinterpret_cast<bf16x4*>(ws + n * 256 + d4 * 4) = o;
        }
      }
      __syncthreads();
      #pragma unroll
      for (int ks = 0; ks < 8; ++ks) {
        bf16x16 a = load_fragA(hs + r0 * kH0 + kt * 256 + ks * 32, kH0, lane);
        #pragma unroll
        for (int nt = 0; nt < 4; ++nt) {
          bf16x16 bm = load_fragB(ws + nt * 16 * 256 + ks * 32, 256, lane);
          acc[nt] = __builtin_amdgcn_wmma_f32_16x16x32_bf16(
              false, a, false, bm, (short)0, acc[nt], false, false);
        }
      }
      __syncthreads();   // ws reuse
    }
    const int mrow = r0 + ((lane >> 4) << 3);
    #pragma unroll
    for (int nt = 0; nt < 4; ++nt) {
      const int col  = nc * 64 + nt * 16 + (lane & 15);
      const float bias = b1g[col];
      #pragma unroll
      for (int v = 0; v < 8; ++v) {
        float val = acc[nt][v] + bias;
        h1s[(mrow + v) * kH1 + col] = f2bf(val > 0.f ? val : 0.f);
      }
    }
  }
  __syncthreads();

  // ===== Layer 2: out[b,c] = relu(h1) . W2[c] + b2[c]  (0.1% of FLOPs) =====
  if (tid < kMT) {
    const float* w2g = W2 + (size_t)c * kH1;     // uniform -> scalar loads
    float accv = 0.f;
    for (int k = 0; k < kH1; k += 8) {
      uint4 hv = *reinterpret_cast<const uint4*>(h1s + tid * kH1 + k);
      const unsigned w[4] = { hv.x, hv.y, hv.z, hv.w };
      #pragma unroll
      for (int j = 0; j < 4; ++j) {
        float lo = __builtin_bit_cast(float, w[j] << 16);          // bf16 -> f32
        float hi = __builtin_bit_cast(float, w[j] & 0xFFFF0000u);
        accv += lo * w2g[k + 2 * j] + hi * w2g[k + 2 * j + 1];
      }
    }
    out[(size_t)(row0 + tid) * kC + c] = accv + b2[c];
  }
}

extern "C" void kernel_launch(void* const* d_in, const int* in_sizes, int n_in,
                              void* d_out, int out_size, void* d_ws, size_t ws_size,
                              hipStream_t stream) {
  (void)in_sizes; (void)n_in; (void)out_size;
  const float* x  = (const float*)d_in[0];
  const float* W0 = (const float*)d_in[1];
  const float* b0 = (const float*)d_in[2];
  const float* W1 = (const float*)d_in[3];
  const float* b1 = (const float*)d_in[4];
  const float* W2 = (const float*)d_in[5];
  const float* b2 = (const float*)d_in[6];
  float* out = (float*)d_out;

  dim3 grid(kBatch / kMT, kC);                              // 128 x 8 = 1024 WGs
  size_t lds_bytes = (size_t)SMEM_ELEMS * sizeof(__bf16);   // 288 KB / WG

  const bool pre = (d_ws != nullptr) && (ws_size >= WS_NEEDED);  // same every call
  if (pre) {
    char* wsb = (char*)d_ws;
    __bf16* xb  = (__bf16*)wsb;
    __bf16* w0b = (__bf16*)(wsb + XB_ELEMS * 2);
    __bf16* w1b = (__bf16*)(wsb + (XB_ELEMS + W0B_ELEMS) * 2);
    const int n4x  = (int)(XB_ELEMS  / 4);
    const int n4w0 = (int)(W0B_ELEMS / 4);
    const int n4w1 = (int)(W1B_ELEMS / 4);
    cvt_f32_to_bf16<<<(n4x  + 255) / 256, 256, 0, stream>>>(x,  xb,  n4x);
    cvt_f32_to_bf16<<<(n4w0 + 255) / 256, 256, 0, stream>>>(W0, w0b, n4w0);
    cvt_f32_to_bf16<<<(n4w1 + 255) / 256, 256, 0, stream>>>(W1, w1b, n4w1);
    ConfounderStackLayers_kernel<true><<<grid, kThreads, lds_bytes, stream>>>(
        xb, w0b, b0, w1b, b1, W2, b2, out);
  } else {
    ConfounderStackLayers_kernel<false><<<grid, kThreads, lds_bytes, stream>>>(
        x, W0, b0, W1, b1, W2, b2, out);
  }
}
